// CConv_38233798869210
// MI455X (gfx1250) — matrix operand order repaired and checked by reading, not compiled, not run on hardware
//
#include <hip/hip_runtime.h>

typedef __attribute__((ext_vector_type(8)))  _Float16 v8h;
typedef __attribute__((ext_vector_type(16))) _Float16 v16h;
typedef __attribute__((ext_vector_type(8)))  float    v8f;

#define CH      128
#define SPA     27
#define MXS     32
#define KTOT    (SPA*CH)        // 3456
#define KSTEPS  (KTOT/32)       // 108
#define NPTS    32              // points per block
#define NWAVES  8
#define TPAD    (KTOT+8)        // 3464 halves per tT row (bank pad, keeps 16B align)
#define PPAD    40              // 80B row stride
#define SPAD    40

static __device__ __forceinline__ v16h cat8(v8h a, v8h b) {
  return __builtin_shufflevector(a, b, 0,1,2,3,4,5,6,7,8,9,10,11,12,13,14,15);
}

// --- prep: feat_in fp32 -> f16 ---
__global__ void feat_to_h(const float* __restrict__ src, _Float16* __restrict__ dst, int total) {
  int t = blockIdx.x * blockDim.x + threadIdx.x;
  if (t < total) dst[t] = (_Float16)src[t];
}

// --- prep: weight[s][o*128+i] fp32 -> WB[o][s*128+i] f16 ---
__global__ void wb_build(const float* __restrict__ src, _Float16* __restrict__ dst, int total) {
  int t = blockIdx.x * blockDim.x + threadIdx.x;
  if (t >= total) return;
  int o = t / KTOT;
  int r = t - o * KTOT;
  int s = r >> 7;         // /128
  int i = r & 127;
  dst[t] = (_Float16)src[(size_t)s * (CH*CH) + o * CH + i];
}

// --- fused gather + stage A + stage B ---
__global__ __launch_bounds__(256) void cconv_main(
    const float*    __restrict__ sel,     // [N,32,27]
    const int*      __restrict__ nidx,    // [N,32]
    const _Float16* __restrict__ feat_h,  // [N,128]
    const _Float16* __restrict__ wb,      // [128][3456]
    float*          __restrict__ out,     // [N,128]
    int npts)
{
  __shared__ alignas(16) _Float16 tT[NPTS][TPAD];        // 221,696 B : t, per-point K-major
  __shared__ alignas(16) _Float16 pT[NWAVES][CH][PPAD];  //  81,920 B : patch^T per wave [i][m]
  __shared__ alignas(16) _Float16 sT[NWAVES][32][SPAD];  //  20,480 B : sel^T per wave [s][m]

  const int tid  = threadIdx.x;
  const int w    = tid >> 5;
  const int lane = tid & 31;
  const int lo   = lane & 15;
  const int hi   = lane >> 4;
  const int n0   = blockIdx.x * NPTS;

  // ================= Stage A: per-point t = sel^T x patch =================
  for (int j = 0; j < 4; ++j) {
    const int p = w * 4 + j;
    const int nr = n0 + p;
    const int n  = (nr < npts) ? nr : (npts - 1);   // clamp: keep EXEC all-ones

    // sel^T into LDS (coalesced reads, f16 convert); pad rows 27..31 with 0
    const float* selp = sel + (size_t)n * (MXS * SPA);
    #pragma unroll
    for (int t = 0; t < 27; ++t) {
      int flat = t * 32 + lane;            // 0..863 == m*27+s
      int m = flat / 27;
      int s = flat - m * 27;
      sT[w][s][m] = (_Float16)selp[flat];
    }
    #pragma unroll
    for (int s = 27; s < 32; ++s) sT[w][s][lane] = (_Float16)0.f;

    // gather neighbor feature rows (f16, L2-resident), store transposed [i][m]
    const int nb = nidx[n * MXS + lane];                 // lane == m
    const _Float16* frow = feat_h + (size_t)nb * CH;
    #pragma unroll
    for (int i0 = 0; i0 < CH; i0 += 8) {
      v8h v = *reinterpret_cast<const v8h*>(frow + i0);  // global_load_b128
      #pragma unroll
      for (int e = 0; e < 8; ++e) pT[w][i0 + e][lane] = v[e];
    }

    __builtin_amdgcn_wave_barrier();  // per-wave buffers; LDS pipe is in-order per wave

    #pragma unroll
    for (int st = 0; st < 2; ++st) {
      // A fragment (16x32 f16): lane row = s, per-lane K chunks {0-7,16-23}/{8-15,24-31}
      const _Float16* ar = &sT[w][st * 16 + lo][hi * 8];
      v16h A = cat8(*reinterpret_cast<const v8h*>(ar),
                    *reinterpret_cast<const v8h*>(ar + 16));
      #pragma unroll
      for (int it = 0; it < 8; ++it) {
        // B fragment (32x16 f16): lane col = i, per-lane K(m) chunks {0-7,16-23}/{8-15,24-31}
        const _Float16* br = &pT[w][it * 16 + lo][hi * 8];
        v16h B = cat8(*reinterpret_cast<const v8h*>(br),
                      *reinterpret_cast<const v8h*>(br + 16));
        v8f C = {};
        C = __builtin_amdgcn_wmma_f32_16x16x32_f16(false, A, false, B,
                                                   (short)0, C, false, false);
        // D: VGPR r -> row s = st*16 + hi*8 + r ; lane -> col i = it*16 + lo
        #pragma unroll
        for (int r = 0; r < 8; ++r) {
          const int srow = st * 16 + hi * 8 + r;
          if (srow < SPA)
            tT[p][srow * CH + it * 16 + lo] = (_Float16)C[r];
        }
      }
    }
    __builtin_amdgcn_wave_barrier();
  }

  __syncthreads();

  // ======= Stage B: out[32 pts][128] = tT x WB^T =======
  // wave w owns output col-tile w (one B-frag stream) x two point row-tiles
  const int o0 = w * 16 + lo;
  const _Float16* wp  = wb + (size_t)o0 * KTOT + hi * 8;
  const _Float16* tr0 = &tT[lo][hi * 8];
  const _Float16* tr1 = &tT[16 + lo][hi * 8];

  v8f acc0 = {}; v8f acc1 = {};
  #pragma unroll 4
  for (int ks = 0; ks < KSTEPS; ++ks) {
    const int k0 = ks * 32;
    v16h B  = cat8(*reinterpret_cast<const v8h*>(wp + k0),        // global_load_b128 (L2)
                   *reinterpret_cast<const v8h*>(wp + k0 + 16));
    v16h A0 = cat8(*reinterpret_cast<const v8h*>(tr0 + k0),       // ds_load_b128
                   *reinterpret_cast<const v8h*>(tr0 + k0 + 16));
    v16h A1 = cat8(*reinterpret_cast<const v8h*>(tr1 + k0),
                   *reinterpret_cast<const v8h*>(tr1 + k0 + 16));
    acc0 = __builtin_amdgcn_wmma_f32_16x16x32_f16(false, A0, false, B,
                                                  (short)0, acc0, false, false);
    acc1 = __builtin_amdgcn_wmma_f32_16x16x32_f16(false, A1, false, B,
                                                  (short)0, acc1, false, false);
  }

  // D: VGPR r -> point p = rowtile*16 + hi*8 + r ; lane -> col o = w*16 + lo
  #pragma unroll
  for (int r = 0; r < 8; ++r) {
    const int p0 = hi * 8 + r;
    if (n0 + p0 < npts)      out[(size_t)(n0 + p0) * CH + o0]      = acc0[r];
    const int p1 = 16 + p0;
    if (n0 + p1 < npts)      out[(size_t)(n0 + p1) * CH + o0]      = acc1[r];
  }
}

extern "C" void kernel_launch(void* const* d_in, const int* in_sizes, int n_in,
                              void* d_out, int out_size, void* d_ws, size_t ws_size,
                              hipStream_t stream) {
  const float* feat = (const float*)d_in[0];
  const float* sel  = (const float*)d_in[1];
  const float* wgt  = (const float*)d_in[2];
  const int*   nidx = (const int*)d_in[3];
  float* out = (float*)d_out;

  const int n_feat = in_sizes[0];      // N*128
  const int npts   = n_feat / CH;      // N

  _Float16* feat_h = (_Float16*)d_ws;                                   // N*128 f16
  _Float16* wbuf   = (_Float16*)((char*)d_ws + (size_t)n_feat * 2);     // 128*3456 f16

  feat_to_h<<<dim3((n_feat + 255) / 256), dim3(256), 0, stream>>>(feat, feat_h, n_feat);
  const int wtot = CH * KTOT;
  wb_build<<<dim3((wtot + 255) / 256), dim3(256), 0, stream>>>(wgt, wbuf, wtot);
  cconv_main<<<dim3((npts + NPTS - 1) / NPTS), dim3(256), 0, stream>>>(
      sel, nidx, feat_h, wbuf, out, npts);
}
